// QueryInteractionModule_72267119723317
// MI455X (gfx1250) — matrix operand order repaired and checked by reading, not compiled, not run on hardware
//
#include <hip/hip_runtime.h>

// ---------------------------------------------------------------------------
// Deformable-DETR decoder (3 layers) for MI455X / gfx1250.
// All dense GEMMs run on v_wmma_f32_16x16x32_bf16 (f32 accumulate), data kept
// f32 in HBM (bandwidth-bound problem), converted to bf16 in registers.
// GEMM uses a 64x32 per-wave macro-tile (4x2 register blocking -> 8 static
// WMMAs per K-step, 4x B-fragment reuse, 2x A-fragment reuse) and
// global_prefetch on the A stream.
// ---------------------------------------------------------------------------

typedef __attribute__((ext_vector_type(16))) __bf16 v16bf;
typedef __attribute__((ext_vector_type(8)))  float  v8f;

// Problem constants
constexpr int B_   = 8;
constexpr int Q_   = 300;
constexpr int D_   = 256;
constexpr int H_   = 8;
constexpr int HD_  = 32;
constexpr int DFF_ = 1024;
constexpr int S_   = 21760;          // 128^2 + 64^2 + 32^2 + 16^2
constexpr int M_TGT = B_ * Q_;       // 2400
constexpr int M_MEM = B_ * S_;       // 174080

static __device__ __forceinline__ v16bf cvt16(float4 a, float4 b, float4 c, float4 d)
{
    v16bf v;
    v[0]  = (__bf16)a.x; v[1]  = (__bf16)a.y; v[2]  = (__bf16)a.z; v[3]  = (__bf16)a.w;
    v[4]  = (__bf16)b.x; v[5]  = (__bf16)b.y; v[6]  = (__bf16)b.z; v[7]  = (__bf16)b.w;
    v[8]  = (__bf16)c.x; v[9]  = (__bf16)c.y; v[10] = (__bf16)c.z; v[11] = (__bf16)c.w;
    v[12] = (__bf16)d.x; v[13] = (__bf16)d.y; v[14] = (__bf16)d.z; v[15] = (__bf16)d.w;
    return v;
}

// C[M,N] = A[M,K] @ W[N,K]^T + bias, optional ReLU.
// One wave computes a 64x32 macro-tile = 4 (M) x 2 (N) WMMA tiles.
// Out-of-range tiles are clamped to the last valid tile (loads stay in
// bounds, EXEC stays all-ones for WMMA); only stores are guarded
// (wave-uniformly).  M % 16 == 0, N % 16 == 0, K % 32 == 0 at every call.
__global__ void gemm_wmma_bf16(const float* __restrict__ A,
                               const float* __restrict__ W,
                               const float* __restrict__ bias,
                               float* __restrict__ C,
                               int M, int N, int K, int relu)
{
    const int wid    = blockIdx.x * (blockDim.x >> 5) + (threadIdx.x >> 5);
    const int tilesM = M >> 4;
    const int tilesN = N >> 4;
    const int mblk   = (tilesM + 3) >> 2;        // macro-tiles along M
    const int nblk   = (tilesN + 1) >> 1;        // macro-tiles along N
    if (wid >= mblk * nblk) return;              // wave-uniform early out

    const int tmb = wid / nblk;
    const int tnb = wid % nblk;

    const int lane   = threadIdx.x & 31;
    const int laneHi = lane >> 4;                // 0|1
    const int laneLo = lane & 15;

    // Clamped tile indices (keep all loads in bounds, EXEC all-ones).
    int tmv[4], tnv[2];
    const float* Abase[4];
    const float* Wbase[2];
#pragma unroll
    for (int mt = 0; mt < 4; ++mt) {
        int tm = tmb * 4 + mt;
        tmv[mt] = tm;
        if (tm > tilesM - 1) tm = tilesM - 1;
        // 16-bit A 16x32 layout: lane<16 holds K {0..7,16..23}, lane>=16 {8..15,24..31}
        Abase[mt] = A + (size_t)(tm * 16 + laneLo) * K + laneHi * 8;
    }
#pragma unroll
    for (int nt = 0; nt < 2; ++nt) {
        int tn = tnb * 2 + nt;
        tnv[nt] = tn;
        if (tn > tilesN - 1) tn = tilesN - 1;
        // 16-bit B 32x16 layout: lane<16 holds K {0..7,16..23} of its column, lane>=16 {8..15,24..31}
        Wbase[nt] = W + (size_t)(tn * 16 + laneLo) * K + laneHi * 8;
    }

    v8f c[4][2];
#pragma unroll
    for (int mt = 0; mt < 4; ++mt)
#pragma unroll
        for (int nt = 0; nt < 2; ++nt)
            c[mt][nt] = (v8f){};

    for (int k0 = 0; k0 < K; k0 += 32) {
        v16bf a[4], b[2];
#pragma unroll
        for (int mt = 0; mt < 4; ++mt) {
            const float4* ap0 = reinterpret_cast<const float4*>(Abase[mt] + k0);
            const float4* ap1 = reinterpret_cast<const float4*>(Abase[mt] + k0 + 16);
            a[mt] = cvt16(ap0[0], ap0[1], ap1[0], ap1[1]);
            __builtin_prefetch(Abase[mt] + k0 + 64, 0, 3);   // global_prefetch_b8
        }
#pragma unroll
        for (int nt = 0; nt < 2; ++nt) {
            const float4* bp = reinterpret_cast<const float4*>(Wbase[nt] + k0);
            b[nt] = cvt16(bp[0], bp[1], bp[4], bp[5]);
        }
#pragma unroll
        for (int mt = 0; mt < 4; ++mt)
#pragma unroll
            for (int nt = 0; nt < 2; ++nt)
                c[mt][nt] = __builtin_amdgcn_wmma_f32_16x16x32_bf16(
                        /*neg_a=*/false, a[mt], /*neg_b=*/false, b[nt],
                        /*c_mod=*/(short)0, c[mt][nt],
                        /*reuse_a=*/false, /*reuse_b=*/false);
    }

    // Epilogue: bias (+ReLU) and guarded stores.
#pragma unroll
    for (int nt = 0; nt < 2; ++nt) {
        if (tnv[nt] >= tilesN) continue;                 // wave-uniform
        const int col = tnv[nt] * 16 + laneLo;
        const float bv = bias[col];
#pragma unroll
        for (int mt = 0; mt < 4; ++mt) {
            if (tmv[mt] >= tilesM) continue;             // wave-uniform
#pragma unroll
            for (int r = 0; r < 8; ++r) {
                const int row = tmv[mt] * 16 + laneHi * 8 + r;  // C layout: VGPR r -> M=r|M=8+r
                float v = c[mt][nt][r] + bv;
                if (relu) v = fmaxf(v, 0.0f);
                C[(size_t)row * N + col] = v;
            }
        }
    }
}

// Self-attention: one wave per (b,h,q); lane owns one head-dim; online softmax.
__global__ void self_attn_kernel(const float* __restrict__ qkv,   // [B*Q, 768]
                                 float* __restrict__ out)         // [B*Q, 256]
{
    const int wid = blockIdx.x * (blockDim.x >> 5) + (threadIdx.x >> 5);
    if (wid >= B_ * H_ * Q_) return;
    const int q = wid % Q_;
    const int h = (wid / Q_) % H_;
    const int b = wid / (Q_ * H_);
    const int d = threadIdx.x & 31;

    const float scale = 0.17677669529663687f;     // 1/sqrt(32)
    const float qv = qkv[(size_t)(b * Q_ + q) * 768 + h * HD_ + d];

    float m = -1e30f, l = 0.0f, acc = 0.0f;
    for (int k = 0; k < Q_; ++k) {
        const size_t base = (size_t)(b * Q_ + k) * 768 + h * HD_ + d;
        float s = qv * qkv[base + 256];
#pragma unroll
        for (int o = 16; o > 0; o >>= 1) s += __shfl_xor(s, o, 32);
        s *= scale;
        const float mn   = fmaxf(m, s);
        const float corr = __expf(m - mn);
        const float e    = __expf(s - mn);
        l   = l * corr + e;
        acc = acc * corr + e * qkv[base + 512];
        m = mn;
    }
    out[(size_t)(b * Q_ + q) * D_ + h * HD_ + d] = acc / l;
}

// MS-deformable sampling: one block per (b,q); thread = (head, head-dim).
__global__ void msdeform_kernel(const float* __restrict__ value,   // [B,S,256]
                                const float* __restrict__ off,     // [B*Q,256]
                                const float* __restrict__ aw,      // [B*Q,128]
                                const float* __restrict__ refp,    // [B,Q,2]
                                const float* __restrict__ vr,      // [B,4,2]
                                const int*   __restrict__ shapes,  // [4,2] (H,W)
                                const int*   __restrict__ starts,  // [4]
                                float* __restrict__ out)           // [B*Q,256]
{
    const int bq = blockIdx.x;
    const int b  = bq / Q_;
    const int h  = threadIdx.x >> 5;
    const int d  = threadIdx.x & 31;

    const float* offr = off + (size_t)bq * 256 + h * 32;    // 16 points x (x,y)
    const float* awr  = aw  + (size_t)bq * 128 + h * 16;

    // softmax over the 16 (level,point) attention weights of this head
    float mx = -1e30f;
#pragma unroll
    for (int i = 0; i < 16; ++i) mx = fmaxf(mx, awr[i]);
    float e[16], sum = 0.0f;
#pragma unroll
    for (int i = 0; i < 16; ++i) { e[i] = __expf(awr[i] - mx); sum += e[i]; }
    const float inv = 1.0f / sum;

    const float rx = refp[bq * 2 + 0];
    const float ry = refp[bq * 2 + 1];

    float acc = 0.0f;
    for (int lev = 0; lev < 4; ++lev) {
        const int Hh = shapes[lev * 2 + 0];
        const int Ww = shapes[lev * 2 + 1];
        const int st = starts[lev];
        const float refx = rx * vr[(b * 4 + lev) * 2 + 0];
        const float refy = ry * vr[(b * 4 + lev) * 2 + 1];
#pragma unroll
        for (int p = 0; p < 4; ++p) {
            const int   pi = lev * 4 + p;
            const float lx = refx + offr[pi * 2 + 0] / (float)Ww;
            const float ly = refy + offr[pi * 2 + 1] / (float)Hh;
            const float x  = lx * Ww - 0.5f;
            const float y  = ly * Hh - 0.5f;
            const float x0f = floorf(x), y0f = floorf(y);
            const int   x0  = (int)x0f,  y0  = (int)y0f;
            const float wx1 = x - x0f,   wy1 = y - y0f;
            const float wx0 = 1.0f - wx1, wy0 = 1.0f - wy1;
            const float pw  = e[pi] * inv;

            float s = 0.0f;
#pragma unroll
            for (int c = 0; c < 4; ++c) {
                const int xi = x0 + (c & 1);
                const int yi = y0 + (c >> 1);
                const float w = ((c & 1) ? wx1 : wx0) * ((c >> 1) ? wy1 : wy0);
                const bool valid = (xi >= 0) && (xi < Ww) && (yi >= 0) && (yi < Hh);
                const int xc = min(max(xi, 0), Ww - 1);
                const int yc = min(max(yi, 0), Hh - 1);
                const size_t vidx = ((size_t)b * S_ + st + yc * Ww + xc) * 256 + h * 32 + d;
                s += valid ? w * value[vidx] : 0.0f;
            }
            acc += pw * s;
        }
    }
    out[(size_t)bq * 256 + h * 32 + d] = acc;
}

// out = LayerNorm(res + add) * g + b ; one 256-thread block per row.
__global__ void add_ln_kernel(const float* __restrict__ res,
                              const float* __restrict__ add,
                              const float* __restrict__ g,
                              const float* __restrict__ bb,
                              float* __restrict__ out)
{
    __shared__ float red[256];
    const int row = blockIdx.x;
    const int t   = threadIdx.x;
    const float x = res[(size_t)row * 256 + t] + add[(size_t)row * 256 + t];

    red[t] = x; __syncthreads();
    for (int s = 128; s > 0; s >>= 1) { if (t < s) red[t] += red[t + s]; __syncthreads(); }
    const float mean = red[0] * (1.0f / 256.0f);
    __syncthreads();

    const float dx = x - mean;
    red[t] = dx * dx; __syncthreads();
    for (int s = 128; s > 0; s >>= 1) { if (t < s) red[t] += red[t + s]; __syncthreads(); }
    const float var = red[0] * (1.0f / 256.0f);

    out[(size_t)row * 256 + t] = dx * rsqrtf(var + 1e-5f) * g[t] + bb[t];
}

__global__ void copy_kernel(const float* __restrict__ in, float* __restrict__ out, size_t n)
{
    const size_t i = (size_t)blockIdx.x * blockDim.x + threadIdx.x;
    if (i < n) out[i] = in[i];
}

// ---------------------------------------------------------------------------

extern "C" void kernel_launch(void* const* d_in, const int* in_sizes, int n_in,
                              void* d_out, int out_size, void* d_ws, size_t ws_size,
                              hipStream_t stream)
{
    (void)in_sizes; (void)n_in; (void)out_size; (void)ws_size;

    const float* tgt    = (const float*)d_in[0];
    const float* refp   = (const float*)d_in[1];
    const float* memory = (const float*)d_in[2];
    const float* vr     = (const float*)d_in[3];
    const int*   shapes = (const int*)d_in[4];
    const int*   starts = (const int*)d_in[5];

    // Workspace layout (floats)
    float* ws    = (float*)d_ws;
    size_t o     = 0;
    float* value = ws + o; o += (size_t)M_MEM * D_;     // 44.56M floats
    float* qkv   = ws + o; o += (size_t)M_TGT * 768;
    float* bufA  = ws + o; o += (size_t)M_TGT * DFF_;   // also reused for 256-wide temps
    float* bufB  = ws + o; o += (size_t)M_TGT * D_;
    float* cur   = ws + o; o += (size_t)M_TGT * D_;
    float* offb  = ws + o; o += (size_t)M_TGT * D_;
    float* awb   = ws + o; o += (size_t)M_TGT * 128;

    auto gemm = [&](const float* A, const float* W, const float* bias, float* C,
                    int M, int N, int K, int relu) {
        const int tilesM = M >> 4, tilesN = N >> 4;
        const int waves  = ((tilesM + 3) / 4) * ((tilesN + 1) / 2);
        const int blocks = (waves + 7) / 8;                   // 8 waves / block
        gemm_wmma_bf16<<<dim3(blocks), dim3(256), 0, stream>>>(A, W, bias, C, M, N, K, relu);
    };

    {   // cur = tgt
        const size_t n = (size_t)M_TGT * D_;
        copy_kernel<<<dim3((unsigned)((n + 255) / 256)), dim3(256), 0, stream>>>(tgt, cur, n);
    }

    for (int li = 0; li < 3; ++li) {
        const int pb = 6 + 22 * li;
        const float* sa_in_w  = (const float*)d_in[pb + 0];
        const float* sa_in_b  = (const float*)d_in[pb + 1];
        const float* sa_out_w = (const float*)d_in[pb + 2];
        const float* sa_out_b = (const float*)d_in[pb + 3];
        const float* off_w    = (const float*)d_in[pb + 4];
        const float* off_b    = (const float*)d_in[pb + 5];
        const float* aw_w     = (const float*)d_in[pb + 6];
        const float* aw_b     = (const float*)d_in[pb + 7];
        const float* v_w      = (const float*)d_in[pb + 8];
        const float* v_b      = (const float*)d_in[pb + 9];
        const float* out_w    = (const float*)d_in[pb + 10];
        const float* out_b    = (const float*)d_in[pb + 11];
        const float* l1_w     = (const float*)d_in[pb + 12];
        const float* l1_b     = (const float*)d_in[pb + 13];
        const float* l2_w     = (const float*)d_in[pb + 14];
        const float* l2_b     = (const float*)d_in[pb + 15];
        const float* n1_g     = (const float*)d_in[pb + 16];
        const float* n1_b     = (const float*)d_in[pb + 17];
        const float* n2_g     = (const float*)d_in[pb + 18];
        const float* n2_b     = (const float*)d_in[pb + 19];
        const float* n3_g     = (const float*)d_in[pb + 20];
        const float* n3_b     = (const float*)d_in[pb + 21];

        // --- self attention ---
        gemm(cur, sa_in_w, sa_in_b, qkv, M_TGT, 768, D_, 0);
        self_attn_kernel<<<dim3((B_ * H_ * Q_ + 7) / 8), dim3(256), 0, stream>>>(qkv, bufB);
        gemm(bufB, sa_out_w, sa_out_b, bufA, M_TGT, D_, D_, 0);
        add_ln_kernel<<<dim3(M_TGT), dim3(256), 0, stream>>>(cur, bufA, n1_g, n1_b, cur);

        // --- MS deformable attention ---
        gemm(memory, v_w, v_b, value, M_MEM, D_, D_, 0);      // dominant GEMM (22.8 GFLOP)
        gemm(cur, off_w, off_b, offb, M_TGT, 256, D_, 0);
        gemm(cur, aw_w, aw_b, awb, M_TGT, 128, D_, 0);
        msdeform_kernel<<<dim3(M_TGT), dim3(256), 0, stream>>>(value, offb, awb, refp, vr,
                                                               shapes, starts, bufB);
        gemm(bufB, out_w, out_b, bufA, M_TGT, D_, D_, 0);
        add_ln_kernel<<<dim3(M_TGT), dim3(256), 0, stream>>>(cur, bufA, n2_g, n2_b, cur);

        // --- FFN ---
        gemm(cur, l1_w, l1_b, bufA, M_TGT, DFF_, D_, 1);      // fused ReLU
        gemm(bufA, l2_w, l2_b, bufB, M_TGT, D_, DFF_, 0);
        add_ln_kernel<<<dim3(M_TGT), dim3(256), 0, stream>>>(cur, bufB, n3_g, n3_b, cur);
    }

    {   // d_out = cur
        const size_t n = (size_t)M_TGT * D_;
        copy_kernel<<<dim3((unsigned)((n + 255) / 256)), dim3(256), 0, stream>>>(cur, (float*)d_out, n);
    }
}
